// SCNLinkPredictor_29566554865988
// MI455X (gfx1250) — matrix operand order repaired and checked by reading, not compiled, not run on hardware
//
#include <hip/hip_runtime.h>
#include <hip/hip_bf16.h>

// ---------------- problem constants (match reference) ----------------
#define NNODES 100000
#define DNB    64
#define FDIM   256
#define HDIM   256
#define EDGES  131072
#define TILE_M 128
#define LDA    264   // padded LDS row stride in halves (528B = +4 bank skew/row)

// ---------------- types ----------------
typedef __attribute__((ext_vector_type(16))) __bf16 bf16x16;
typedef __attribute__((ext_vector_type(8)))  __bf16 bf16x8;
typedef __attribute__((ext_vector_type(4)))  __bf16 bf16x4;
typedef __attribute__((ext_vector_type(8)))  float  f32x8;
typedef __attribute__((ext_vector_type(4)))  int    v4i;
typedef __attribute__((ext_vector_type(4)))  unsigned int u32x4;
typedef __attribute__((ext_vector_type(8)))  int    i32x8;
typedef __attribute__((ext_vector_type(4)))  int    i32x4;

union V16 { bf16x16 v; bf16x8 h[2]; };

#if __has_builtin(__builtin_amdgcn_tensor_load_to_lds) && \
    __has_builtin(__builtin_amdgcn_s_wait_tensorcnt)
#define USE_TDM 1
#endif

#if __has_builtin(__builtin_amdgcn_global_load_async_to_lds_b128) && \
    __has_builtin(__builtin_amdgcn_s_wait_asynccnt)
#define USE_ASYNC_LDS 1
#endif

typedef __attribute__((address_space(1))) v4i* glb_v4i_p;   // AS1 == __device__
typedef __attribute__((address_space(3))) v4i* lds_v4i_p;   // AS3 == __shared__

static __device__ inline glb_v4i_p to_glb(const void* p) {
  return (glb_v4i_p)(unsigned long long)(uintptr_t)p;       // generic global == AS1 address
}
static __device__ inline lds_v4i_p to_lds(void* p) {
  return (lds_v4i_p)(unsigned int)(uintptr_t)p;             // low 32 bits == AS3 offset
}

// ---------------- weight prep: f32 [K][N] -> bf16 [N][K] (transposed) ----------------
__global__ __launch_bounds__(256) void prep_weights(
    const float* __restrict__ w0, const float* __restrict__ w1,
    const float* __restrict__ w2, const float* __restrict__ w3,
    const float* __restrict__ w4, __bf16* __restrict__ dst)
{
  const float* srcs[5] = {w0, w1, w2, w3, w4};
  const float* s = srcs[blockIdx.y];
  __bf16* d = dst + (size_t)blockIdx.y * 65536;
  int idx = blockIdx.x * 256 + threadIdx.x;   // 0..65535
  int n = idx >> 8, k = idx & 255;
  d[n * 256 + k] = (__bf16)s[k * 256 + n];
}

// ---------------- stage one 256x256 bf16 weight into LDS (padded stride LDA) ----------------
static __device__ void stage_w(__bf16* wbuf, const __bf16* __restrict__ src) {
#if defined(USE_TDM)
  // One TDM descriptor moves the whole 128KB tile; hardware pad inserts 16B per 512B row,
  // reproducing the LDA=264-half padded layout (pad_interval=6 -> 128 DWORDs, pad_amount=3 -> 4 DWORDs).
  if (threadIdx.x < 32) {            // wave 0 issues the DMA once
    unsigned long long ga = (unsigned long long)(uintptr_t)src;
    unsigned int       la = (unsigned int)(uintptr_t)wbuf;     // LDS byte offset
    u32x4 g0;
    g0.x = 1u;                                                  // count=1 (valid user D#)
    g0.y = la;                                                  // lds_addr  [63:32]
    g0.z = (unsigned int)ga;                                    // global_addr[95:64]
    g0.w = (unsigned int)((ga >> 32) & 0x01FFFFFFu) | (2u << 30); // addr[56:32] | type=2
    i32x8 g1;
    g1[0] = (int)((1u << 16)        // data_size = 1 -> 2 bytes
                | (1u << 20)        // pad_enable
                | (6u << 22)        // pad_interval: 128 DWORDs (512B) before padding
                | (3u << 25));      // pad_amount: 4 DWORDs (16B)
    g1[1] = (int)(256u << 16);      // tensor_dim0 = 256  (bits 79:48, low half)
    g1[2] = (int)(256u << 16);      // tensor_dim1 = 256  (bits 111:80, low half)
    g1[3] = (int)(256u << 16);      // tile_dim0   = 256  (bits 127:112)
    g1[4] = 256;                    // tile_dim1   = 256  (bits 143:128)
    g1[5] = 256;                    // tensor_dim0_stride = 256 (bits 207:160, low 32)
    g1[6] = 0;
    g1[7] = 0;
    i32x4 z4 = {0, 0, 0, 0};        // groups 2/3 unused (2-D tensor)
    i32x8 z8 = {0, 0, 0, 0, 0, 0, 0, 0};
    __builtin_amdgcn_tensor_load_to_lds(g0, g1, z4, z4, z8, 0);   // 6-arg toolchain form
    __builtin_amdgcn_s_wait_tensorcnt(0);
  }
#elif defined(USE_ASYNC_LDS)
  int tid = threadIdx.x;
  for (int idx = tid; idx < 256 * 32; idx += 256) {
    int r = idx >> 5, sgm = idx & 31;                       // 32 x 16B per row
    __builtin_amdgcn_global_load_async_to_lds_b128(
        to_glb(src + r * 256 + sgm * 8),
        to_lds(wbuf + r * LDA + sgm * 8), 0, 0);
  }
  __builtin_amdgcn_s_wait_asynccnt(0);   // this wave's copies done before barrier arrival
#else
  int tid = threadIdx.x;
  for (int idx = tid; idx < 256 * 32; idx += 256) {
    int r = idx >> 5, sgm = idx & 31;
    *(float4*)(wbuf + r * LDA + sgm * 8) = *(const float4*)(src + r * 256 + sgm * 8);
  }
#endif
}

// ---------------- in-place [128,256] x [256,256] GEMM on WMMA bf16 ----------------
static __device__ void gemm256(__bf16* act, const __bf16* wbuf,
                               const float* __restrict__ bias, int relu, float scale,
                               const __bf16* cinit)
{
  const int lane = threadIdx.x & 31;
  const int wv   = threadIdx.x >> 5;     // 8 waves -> 8 M-bands of 16 rows
  const int m0   = wv * 16;
  const int lm   = lane & 15;
  const int hb   = lane >> 4;            // half-wave selector

  // Cache all A fragments for this wave's band (K = 0..255) in registers.
  V16 af[8];
  const int arow = (m0 + lm) * LDA;
#pragma unroll
  for (int kt = 0; kt < 8; ++kt) {
    int ab = arow + kt * 32 + hb * 8;
    af[kt].h[0] = *(const bf16x8*)(act + ab);
    af[kt].h[1] = *(const bf16x8*)(act + ab + 16);
  }

#pragma unroll 1
  for (int nt = 0; nt < 16; ++nt) {
    const int col = nt * 16 + lm;
    f32x8 acc;
    if (cinit) {
#pragma unroll
      for (int r = 0; r < 8; ++r)
        acc[r] = (float)cinit[(m0 + r + hb * 8) * LDA + col];
    } else {
      acc = f32x8{0.f, 0.f, 0.f, 0.f, 0.f, 0.f, 0.f, 0.f};
    }
    const float bv = bias[col];
#pragma unroll
    for (int kt = 0; kt < 8; ++kt) {
      V16 bfg;
      int wb = col * LDA + kt * 32 + hb * 8;
      bfg.h[0] = *(const bf16x8*)(wbuf + wb);
      bfg.h[1] = *(const bf16x8*)(wbuf + wb + 16);
      acc = __builtin_amdgcn_wmma_f32_16x16x32_bf16(false, af[kt].v, false, bfg.v,
                                                    (short)0, acc, false, false);
    }
#pragma unroll
    for (int r = 0; r < 8; ++r) {
      float v = acc[r] + bv;
      if (relu) v = fmaxf(v, 0.f);
      v *= scale;
      act[(m0 + r + hb * 8) * LDA + col] = (__bf16)v;
    }
  }
}

// ---------------- fused per-edge-tile kernel ----------------
__global__ __launch_bounds__(256) void scn_fused(
    const float* __restrict__ x, const int* __restrict__ nbr,
    const int* __restrict__ tar, const float* __restrict__ beta,
    const float* __restrict__ cn_w1, const float* __restrict__ cn_b1,
    const float* __restrict__ cn_b2, const float* __restrict__ cn_b3,
    const float* __restrict__ ij_b1, const float* __restrict__ ij_b2,
    const float* __restrict__ lin_b1, const float* __restrict__ lin_w2,
    const float* __restrict__ lin_b2,
    const __bf16* __restrict__ wts,   // 5x bf16 [N][K] from prep_weights
    float* __restrict__ out)
{
  extern __shared__ __align__(16) unsigned char smem[];
  __bf16* bufA = (__bf16*)smem;                   // cn branch [128][LDA]
  __bf16* bufB = bufA + TILE_M * LDA;             // ij branch [128][LDA]
  __bf16* wbuf = bufB + TILE_M * LDA;             // weights   [256][LDA]
  float* xcn = (float*)(wbuf + 256 * LDA);
  int*   sArr = (int*)(xcn + TILE_M);
  int*   dArr = sArr + TILE_M;

  const int tid = threadIdx.x;
  const int tileBase = blockIdx.x * TILE_M;
  const float betaV = beta[0];

  if (tid < TILE_M) {
    sArr[tid] = tar[tileBase + tid];
    dArr[tid] = tar[EDGES + tileBase + tid];
  }
  __syncthreads();

  // --- common-neighbor pair counts via run-aware merge over sorted lists (L2-resident) ---
  if (tid < TILE_M) {
    const int* na = nbr + (long)sArr[tid] * DNB;
    const int* nb = nbr + (long)dArr[tid] * DNB;
    int ia = 0, ib = 0, cnt = 0;
    while (ia < DNB && ib < DNB) {
      int a = na[ia], b = nb[ib];
      if (a < b) { ia++; }
      else if (b < a) { ib++; }
      else {
        int ca = 0; while (ia < DNB && na[ia] == a) { ca++; ia++; }
        int cb = 0; while (ib < DNB && nb[ib] == a) { cb++; ib++; }
        cnt += ca * cb;    // all-pairs equality semantics under duplicates
      }
    }
    xcn[tid] = (float)cnt;
  }
  __syncthreads();

  // --- bufA = relu(xcn * cn_w1 + cn_b1), bufB = bf16(xi * xj) ---
  for (int idx = tid; idx < TILE_M * HDIM; idx += 256) {
    int e = idx >> 8, h = idx & 255;
    float v = fmaxf(xcn[e] * cn_w1[h] + cn_b1[h], 0.f);
    bufA[e * LDA + h] = (__bf16)v;
  }
  for (int idx = tid; idx < TILE_M * (FDIM / 4); idx += 256) {
    int e = idx >> 6, c = (idx & 63) * 4;
    const float4 a = *(const float4*)(x + (long)sArr[e] * FDIM + c);
    const float4 b = *(const float4*)(x + (long)dArr[e] * FDIM + c);
    bf16x4 o;
    o.x = (__bf16)(a.x * b.x); o.y = (__bf16)(a.y * b.y);
    o.z = (__bf16)(a.z * b.z); o.w = (__bf16)(a.w * b.w);
    *(bf16x4*)(bufB + e * LDA + c) = o;
  }

  // --- cn branch: bufA = beta * (relu(bufA@cn_w2+b2) @ cn_w3 + b3) ---
  __syncthreads();
  stage_w(wbuf, wts + 0 * 65536);          // cn_w2^T
  __syncthreads();
  gemm256(bufA, wbuf, cn_b2, 1, 1.0f, nullptr);
  __syncthreads();
  stage_w(wbuf, wts + 1 * 65536);          // cn_w3^T
  __syncthreads();
  gemm256(bufA, wbuf, cn_b3, 0, betaV, nullptr);

  // --- ij branch: bufB = relu(bufB@ij_w1+b1); then bufB = bufB@ij_w2+b2 + bufA ---
  __syncthreads();
  stage_w(wbuf, wts + 2 * 65536);          // ij_w1^T
  __syncthreads();
  gemm256(bufB, wbuf, ij_b1, 1, 1.0f, nullptr);
  __syncthreads();
  stage_w(wbuf, wts + 3 * 65536);          // ij_w2^T
  __syncthreads();
  gemm256(bufB, wbuf, ij_b2, 0, 1.0f, bufA);   // accumulator seeded with beta*xcn_feat

  // --- lin: bufB = relu(bufB @ lin_w1 + lin_b1) ---
  __syncthreads();
  stage_w(wbuf, wts + 4 * 65536);          // lin_w1^T
  __syncthreads();
  gemm256(bufB, wbuf, lin_b1, 1, 1.0f, nullptr);
  __syncthreads();

  // --- head: out = z @ lin_w2 + lin_b2 (per-edge dot of 256) ---
  if (tid < TILE_M) {
    float s = 0.f;
    const __bf16* row = bufB + tid * LDA;
    for (int c = 0; c < 256; ++c) s += (float)row[c] * lin_w2[c];
    out[tileBase + tid] = s + lin_b2[0];
  }
}

// ---------------- host-side launch ----------------
extern "C" void kernel_launch(void* const* d_in, const int* in_sizes, int n_in,
                              void* d_out, int out_size, void* d_ws, size_t ws_size,
                              hipStream_t stream) {
  const float* x       = (const float*)d_in[0];
  const int*   nbr     = (const int*)d_in[1];
  const int*   tar     = (const int*)d_in[2];
  const float* beta    = (const float*)d_in[3];
  const float* cn_w1   = (const float*)d_in[4];
  const float* cn_b1   = (const float*)d_in[5];
  const float* cn_w2   = (const float*)d_in[6];
  const float* cn_b2   = (const float*)d_in[7];
  const float* cn_w3   = (const float*)d_in[8];
  const float* cn_b3   = (const float*)d_in[9];
  const float* ij_w1   = (const float*)d_in[10];
  const float* ij_b1   = (const float*)d_in[11];
  const float* ij_w2   = (const float*)d_in[12];
  const float* ij_b2   = (const float*)d_in[13];
  const float* lin_w1  = (const float*)d_in[14];
  const float* lin_b1  = (const float*)d_in[15];
  const float* lin_w2  = (const float*)d_in[16];
  const float* lin_b2  = (const float*)d_in[17];

  __bf16* wts = (__bf16*)d_ws;   // 5 * 65536 bf16 = 640 KB

  prep_weights<<<dim3(256, 5), 256, 0, stream>>>(cn_w2, cn_w3, ij_w1, ij_w2, lin_w1, wts);

  const size_t lds = (size_t)(2 * TILE_M * LDA + 256 * LDA) * 2   // bufA+bufB+wbuf (bf16)
                   + TILE_M * 4 + 2 * TILE_M * 4;                 // xcn + sArr/dArr
  scn_fused<<<EDGES / TILE_M, 256, lds, stream>>>(
      x, nbr, tar, beta, cn_w1, cn_b1, cn_b2, cn_b3,
      ij_b1, ij_b2, lin_b1, lin_w2, lin_b2, wts, (float*)d_out);
}
